// PositionAttentionModule_1168231105166
// MI455X (gfx1250) — matrix operand order repaired and checked by reading, not compiled, not run on hardware
//
#include <hip/hip_runtime.h>

typedef __attribute__((ext_vector_type(16))) _Float16     v16h;
typedef __attribute__((ext_vector_type(8)))  float        v8f;
typedef __attribute__((ext_vector_type(8)))  _Float16     v8h;
typedef __attribute__((ext_vector_type(4)))  unsigned int u32x4;

union FragU { v16h h; u32x4 u[2]; };      // 32 bytes: one 16x32 f16 A-frag or 32x16 B-frag per lane
union HalfStoreU { v8h h; u32x4 u; };     // 16 bytes

#define WMMA_F16(a, b, c) \
    __builtin_amdgcn_wmma_f32_16x16x32_f16(false, (a), false, (b), (short)0, (c), false, false)

constexpr int kB  = 8;
constexpr int kC  = 256;
constexpr int kC8 = 32;
constexpr int kN  = 4096;   // 64*64

// ---------------------------------------------------------------------------
// Kernel 0: one-time f32 -> f16 conversion of the (tiny, heavily reused)
// projection weights, so the GEMM kernels load f16 fragments directly.
// ---------------------------------------------------------------------------
__global__ __launch_bounds__(256)
void pam_cvt_w(const float* __restrict__ wq, const float* __restrict__ wk,
               const float* __restrict__ wv,
               _Float16* __restrict__ Wq16, _Float16* __restrict__ Wk16,
               _Float16* __restrict__ Wv16)
{
    const int i = blockIdx.x * 256 + threadIdx.x;
    if (i < kC8 * kC) {
        Wq16[i] = (_Float16)wq[i];
        Wk16[i] = (_Float16)wk[i];
    }
    if (i < kC * kC)
        Wv16[i] = (_Float16)wv[i];
}

// ---------------------------------------------------------------------------
// Kernel 1: fused Q/K/V 1x1-conv projections.
// One wave per 16-row n-tile. D[n][o] = sum_c x[b][c][n] * w[o][c] (+bias).
// A-fragment (Xt) is shared across the Q, K and all 16 V output tiles.
// All 20 W B-fragments are preloaded into registers so the WMMA burst gets
// staggered loadcnt waits instead of a full L2 round-trip per WMMA.
// Q,K stored f16 as [B][N][32]; V stored f16 as [B][C][N].
// ---------------------------------------------------------------------------
__global__ __launch_bounds__(128)
void pam_proj(const float* __restrict__ x,
              const _Float16* __restrict__ Wq16, const float* __restrict__ bq,
              const _Float16* __restrict__ Wk16, const float* __restrict__ bk,
              const _Float16* __restrict__ Wv16, const float* __restrict__ bv,
              _Float16* __restrict__ Qp, _Float16* __restrict__ Kp,
              _Float16* __restrict__ Vp)
{
    const int lane = threadIdx.x & 31;
    const int wave = threadIdx.x >> 5;
    const int tile = blockIdx.x * 4 + wave;      // 0..2047
    const int b    = tile >> 8;
    const int n0   = (tile & 255) << 4;
    const int hl   = lane & 15;
    const int hi   = lane >> 4;                  // 0 or 1

    const float* xb = x + (size_t)b * kC * kN;

    v8f accQ[2], accK[2], accV[16];
    #pragma unroll
    for (int i = 0; i < 2; ++i) { accQ[i] = {}; accK[i] = {}; }
    #pragma unroll
    for (int i = 0; i < 16; ++i) accV[i] = {};

    for (int kc = 0; kc < kC; kc += 32) {
        // A fragment: rows n (M), K = input channel c. Per ISA f16 A layout:
        // lane<16 holds K=[0..8)+[16..24), lane>=16 holds K=[8..16)+[24..32).
        FragU A;
        {
            const int n  = n0 + hl;
            const int c0 = kc + hi * 8;
            #pragma unroll
            for (int j = 0; j < 8; ++j) A.h[j]     = (_Float16)xb[(size_t)(c0 + j) * kN + n];
            #pragma unroll
            for (int j = 0; j < 8; ++j) A.h[8 + j] = (_Float16)xb[(size_t)(c0 + 16 + j) * kN + n];
        }
        // Preload ALL W B-fragments for this k-chunk (2 Q + 2 K + 16 V):
        // column o = lane&15; lane<16 holds K=[0..8)+[16..24), lane>=16 K=[8..16)+[24..32).
        const int cb = kc + hi * 8;
        FragU Wf[20];
        #pragma unroll
        for (int ot = 0; ot < 2; ++ot) {
            const _Float16* wrow = Wq16 + (size_t)(ot * 16 + hl) * kC + cb;
            Wf[ot].u[0] = *reinterpret_cast<const u32x4*>(wrow);
            Wf[ot].u[1] = *reinterpret_cast<const u32x4*>(wrow + 16);
        }
        #pragma unroll
        for (int ot = 0; ot < 2; ++ot) {
            const _Float16* wrow = Wk16 + (size_t)(ot * 16 + hl) * kC + cb;
            Wf[2 + ot].u[0] = *reinterpret_cast<const u32x4*>(wrow);
            Wf[2 + ot].u[1] = *reinterpret_cast<const u32x4*>(wrow + 16);
        }
        #pragma unroll
        for (int ot = 0; ot < 16; ++ot) {
            const _Float16* wrow = Wv16 + (size_t)(ot * 16 + hl) * kC + cb;
            Wf[4 + ot].u[0] = *reinterpret_cast<const u32x4*>(wrow);
            Wf[4 + ot].u[1] = *reinterpret_cast<const u32x4*>(wrow + 16);
        }
        // WMMA burst: each op only waits for its own (already in-flight) pair.
        #pragma unroll
        for (int ot = 0; ot < 2; ++ot)
            accQ[ot] = WMMA_F16(A.h, Wf[ot].h, accQ[ot]);
        #pragma unroll
        for (int ot = 0; ot < 2; ++ot)
            accK[ot] = WMMA_F16(A.h, Wf[2 + ot].h, accK[ot]);
        #pragma unroll
        for (int ot = 0; ot < 16; ++ot)
            accV[ot] = WMMA_F16(A.h, Wf[4 + ot].h, accV[ot]);
    }

    // D layout: lane holds column o = lane&15, rows n = n0 + hi*8 + r.
    _Float16* Qb = Qp + (size_t)b * kN * kC8;
    _Float16* Kb = Kp + (size_t)b * kN * kC8;
    _Float16* Vb = Vp + (size_t)b * kC * kN;

    #pragma unroll
    for (int ot = 0; ot < 2; ++ot) {
        const int o = ot * 16 + hl;
        const float biq = bq[o], bik = bk[o];
        #pragma unroll
        for (int r = 0; r < 8; ++r) {
            const int n = n0 + hi * 8 + r;
            Qb[(size_t)n * kC8 + o] = (_Float16)(accQ[ot][r] + biq);
            Kb[(size_t)n * kC8 + o] = (_Float16)(accK[ot][r] + bik);
        }
    }
    #pragma unroll
    for (int ot = 0; ot < 16; ++ot) {
        const int c = ot * 16 + hl;
        const float biv = bv[c];
        HalfStoreU st;
        #pragma unroll
        for (int r = 0; r < 8; ++r) st.h[r] = (_Float16)(accV[ot][r] + biv);
        *reinterpret_cast<u32x4*>(Vb + (size_t)c * kN + n0 + hi * 8) = st.u;
    }
}

// ---------------------------------------------------------------------------
// Kernel 2: fused flash-style attention + residual.
// One wave per 16-row query tile.
//   Pass 1: row MAX only (no exp), 4 energy tiles in flight per iteration.
//   Pass 2: recompute energy, P = exp(e-max) -> LDS -> A-frag; all 16 V
//           fragments are loaded into registers BEFORE the WMMA burst so
//           loads pipeline (staggered loadcnt waits) instead of serializing.
// Exact softmax: normalization 1/sum applied in the epilogue.
// ---------------------------------------------------------------------------
__global__ __launch_bounds__(128)
void pam_attn(const float* __restrict__ x,
              const _Float16* __restrict__ Qp, const _Float16* __restrict__ Kp,
              const _Float16* __restrict__ Vp, float* __restrict__ out)
{
    __shared__ _Float16 plds[4][16 * 32];    // 1 KB P-staging tile per wave

    const int lane = threadIdx.x & 31;
    const int wave = threadIdx.x >> 5;
    const int tile = blockIdx.x * 4 + wave;  // 0..2047
    const int b    = tile >> 8;
    const int n0   = (tile & 255) << 4;
    const int hl   = lane & 15;
    const int hi   = lane >> 4;

    const _Float16* Qb = Qp + (size_t)b * kN * kC8;
    const _Float16* Kb = Kp + (size_t)b * kN * kC8;
    const _Float16* Vb = Vp + (size_t)b * kC * kN;

    // Persistent Q A-fragment (rows n, K = c8, reused for all 4096 m).
    FragU Qa;
    {
        const _Float16* qrow = Qb + (size_t)(n0 + hl) * kC8 + hi * 8;
        Qa.u[0] = *reinterpret_cast<const u32x4*>(qrow);
        Qa.u[1] = *reinterpret_cast<const u32x4*>(qrow + 16);
    }
    // Base for K B-fragment loads: column m = lane&15, K rows split by half.
    const _Float16* kbase = Kb + (size_t)hl * kC8 + hi * 8;

    // -------- Pass 1: row max only (matrix-op bound, no exp) ---------------
    float mx[8];
    #pragma unroll
    for (int r = 0; r < 8; ++r) mx[r] = -3.0e38f;

    for (int m0 = 0; m0 < kN; m0 += 64) {
        FragU Kf[4];
        #pragma unroll
        for (int t = 0; t < 4; ++t) {
            const _Float16* krow = kbase + (size_t)(m0 + t * 16) * kC8;
            Kf[t].u[0] = *reinterpret_cast<const u32x4*>(krow);
            Kf[t].u[1] = *reinterpret_cast<const u32x4*>(krow + 16);
        }
        v8f e[4];
        #pragma unroll
        for (int t = 0; t < 4; ++t) {
            v8f cz = {};
            e[t] = WMMA_F16(Qa.h, Kf[t].h, cz);
        }
        #pragma unroll
        for (int r = 0; r < 8; ++r)
            mx[r] = fmaxf(fmaxf(fmaxf(mx[r], e[0][r]), fmaxf(e[1][r], e[2][r])), e[3][r]);
    }
    // Row n = hi*8 + r lives across one 16-lane half: butterfly max merge.
    #pragma unroll
    for (int s = 1; s < 16; s <<= 1) {
        #pragma unroll
        for (int r = 0; r < 8; ++r)
            mx[r] = fmaxf(mx[r], __shfl_xor(mx[r], s, 16));
    }

    // -------- Pass 2: O[n][c] accumulation + raw row-sum -------------------
    v8f O[16];
    #pragma unroll
    for (int i = 0; i < 16; ++i) O[i] = {};
    float sm[8];
    #pragma unroll
    for (int r = 0; r < 8; ++r) sm[r] = 0.f;

    _Float16* Pl = plds[wave];

    for (int m0 = 0; m0 < kN; m0 += 32) {
        // Energy for this 16x32 m-chunk (two tiles).
        v8f e0, e1;
        {
            FragU Kf[2];
            #pragma unroll
            for (int t = 0; t < 2; ++t) {
                const _Float16* krow = kbase + (size_t)(m0 + t * 16) * kC8;
                Kf[t].u[0] = *reinterpret_cast<const u32x4*>(krow);
                Kf[t].u[1] = *reinterpret_cast<const u32x4*>(krow + 16);
            }
            v8f cz0 = {};
            e0 = WMMA_F16(Qa.h, Kf[0].h, cz0);
            v8f cz1 = {};
            e1 = WMMA_F16(Qa.h, Kf[1].h, cz1);
        }

        // Issue ALL 16 V B-fragment loads up front: their latency overlaps
        // the exp/LDS transpose below, and the WMMA burst gets staggered
        // loadcnt waits instead of per-tile full serialization.
        FragU Vf[16];
        #pragma unroll
        for (int ct = 0; ct < 16; ++ct) {
            const _Float16* vrow = Vb + (size_t)(ct * 16 + hl) * kN + m0 + hi * 8;
            Vf[ct].u[0] = *reinterpret_cast<const u32x4*>(vrow);
            Vf[ct].u[1] = *reinterpret_cast<const u32x4*>(vrow + 16);
        }

        // P = exp(e - rowmax) as f16, staged row-major [16][32] in LDS;
        // accumulate the (unnormalized) row sum on the fly.
        #pragma unroll
        for (int r = 0; r < 8; ++r) {
            const int row = hi * 8 + r;
            const float p0 = __expf(e0[r] - mx[r]);
            const float p1 = __expf(e1[r] - mx[r]);
            sm[r] += p0 + p1;
            Pl[row * 32 + hl]      = (_Float16)p0;
            Pl[row * 32 + 16 + hl] = (_Float16)p1;
        }
        asm volatile("s_wait_dscnt 0" ::: "memory");
        // Re-read as A-fragment (layout transpose C/D -> A via LDS).
        FragU Pa;
        {
            const _Float16* prow = Pl + hl * 32 + hi * 8;
            Pa.u[0] = *reinterpret_cast<const u32x4*>(prow);
            Pa.u[1] = *reinterpret_cast<const u32x4*>(prow + 16);
        }
        // O[n][c] += P(16x32) * V^T(32x16) over all 16 c-tiles.
        #pragma unroll
        for (int ct = 0; ct < 16; ++ct)
            O[ct] = WMMA_F16(Pa.h, Vf[ct].h, O[ct]);
    }

    // Cross-lane (16-wide) sum reduction, once per tile.
    #pragma unroll
    for (int s = 1; s < 16; s <<= 1) {
        #pragma unroll
        for (int r = 0; r < 8; ++r)
            sm[r] += __shfl_xor(sm[r], s, 16);
    }
    float rs[8];
    #pragma unroll
    for (int r = 0; r < 8; ++r) rs[r] = 1.0f / sm[r];

    // -------- Epilogue: scale by 1/rowsum, add residual, store -------------
    const float* xb = x   + (size_t)b * kC * kN;
    float*       ob = out + (size_t)b * kC * kN;
    #pragma unroll
    for (int ct = 0; ct < 16; ++ct) {
        const int c = ct * 16 + hl;
        const size_t base = (size_t)c * kN + n0 + hi * 8;
        #pragma unroll
        for (int r = 0; r < 8; ++r)
            ob[base + r] = O[ct][r] * rs[r] + xb[base + r];
    }
}

// ---------------------------------------------------------------------------
extern "C" void kernel_launch(void* const* d_in, const int* in_sizes, int n_in,
                              void* d_out, int out_size, void* d_ws, size_t ws_size,
                              hipStream_t stream) {
    const float* x  = (const float*)d_in[0];
    const float* wq = (const float*)d_in[1];
    const float* bq = (const float*)d_in[2];
    const float* wk = (const float*)d_in[3];
    const float* bk = (const float*)d_in[4];
    const float* wv = (const float*)d_in[5];
    const float* bv = (const float*)d_in[6];
    float* out = (float*)d_out;

    char* ws = (char*)d_ws;
    const size_t qkBytes = (size_t)kB * kN * kC8 * sizeof(_Float16);  // 2 MB each
    const size_t vBytes  = (size_t)kB * kC * kN * sizeof(_Float16);   // 16 MB
    _Float16* Qp = (_Float16*)(ws);
    _Float16* Kp = (_Float16*)(ws + qkBytes);
    _Float16* Vp = (_Float16*)(ws + 2 * qkBytes);
    char* wsW = ws + 2 * qkBytes + vBytes;
    _Float16* Wq16 = (_Float16*)(wsW);                                // 16 KB
    _Float16* Wk16 = (_Float16*)(wsW + (size_t)kC8 * kC * 2);         // 16 KB
    _Float16* Wv16 = (_Float16*)(wsW + 2 * (size_t)kC8 * kC * 2);     // 128 KB

    const int tiles  = kB * (kN / 16);   // 2048 query tiles
    const int blocks = tiles / 4;        // 4 waves (one tile each) per 128-thread block

    pam_cvt_w<<<(kC * kC) / 256, 256, 0, stream>>>(wq, wk, wv, Wq16, Wk16, Wv16);
    pam_proj<<<blocks, 128, 0, stream>>>(x, Wq16, bq, Wk16, bk, Wv16, bv, Qp, Kp, Vp);
    pam_attn<<<blocks, 128, 0, stream>>>(x, Qp, Kp, Vp, out);
}
